// HyenaOperator_68556267978771
// MI455X (gfx1250) — matrix operand order repaired and checked by reading, not compiled, not run on hardware
//
#include <hip/hip_runtime.h>
#include <hip/hip_bf16.h>

// ---------------------------------------------------------------------------
// Hyena forward for MI455X (gfx1250, wave32, WMMA).
// B=2, L=8192, C=1024, ORDER=2.  Only order index 1 survives the reference
// loop: out = (x2 * irfft(rfft(shortconv(v)) * rfft(h1))) @ W_out + b_out
// ---------------------------------------------------------------------------

typedef _Float16 half16 __attribute__((ext_vector_type(16)));
typedef _Float16 half8  __attribute__((ext_vector_type(8)));
typedef float    f32x8  __attribute__((ext_vector_type(8)));
typedef int      i32x4  __attribute__((ext_vector_type(4)));

static constexpr int Bb   = 2;
static constexpr int Ls   = 8192;
static constexpr int Cs   = 1024;
static constexpr int Mrow = Bb * Ls;        // 16384
static constexpr int Nfft = 2 * Ls;         // 16384
static constexpr int LGN  = 14;

#if __has_builtin(__builtin_amdgcn_global_load_async_to_lds_b128) && \
    __has_builtin(__builtin_amdgcn_s_wait_asynccnt)
#define USE_ASYNC_LDS 1
#else
#define USE_ASYNC_LDS 0
#endif

#define AS_GLOBAL __attribute__((address_space(1)))
#define AS_LDS    __attribute__((address_space(3)))

// ---------------- fp32 -> fp16 convert (A operand, row-major kept) ----------------
__global__ void cvt_f16_kernel(const float* __restrict__ in, _Float16* __restrict__ out, int n) {
    int i = blockIdx.x * blockDim.x + threadIdx.x;
    if (i < n) out[i] = (_Float16)in[i];
}

// ---------------- fp32 (R,C) -> f16 (C,R) transpose-convert (weights) ----------------
__global__ __launch_bounds__(256)
void cvt_transpose_f16_kernel(const float* __restrict__ in, _Float16* __restrict__ out,
                              int R, int C)
{
    __shared__ float tile[32][33];
    int r0 = blockIdx.x * 32, c0 = blockIdx.y * 32;
    for (int rr = threadIdx.y; rr < 32; rr += 8)
        tile[rr][threadIdx.x] = in[(size_t)(r0 + rr) * C + c0 + threadIdx.x];
    __syncthreads();
    for (int rr = threadIdx.y; rr < 32; rr += 8)
        out[(size_t)(c0 + rr) * R + r0 + threadIdx.x] = (_Float16)tile[threadIdx.x][rr];
}

// ---------------- WMMA GEMM: C = A(f16,MxK) @ Bt(f16,NxK)^T + bias ----------------
// Both operands row-major along K -> identical contiguous 16B tile staging.
// Double-buffered LDS, async global->LDS (ASYNCcnt) when available.
// MODE 0: out0[row*N+col] = val
// MODE 1: split-proj: slot=col%3, c=col/3 ; slot0->out0 (v_pre), slot2->out1 (x2)
static constexpr int BM = 128, BN = 128, BK = 64, LPAD = 8, BKP = BK + LPAD;

template<int MODE>
__global__ __launch_bounds__(256)
void wmma_gemm_kernel(const _Float16* __restrict__ A, const _Float16* __restrict__ Bt,
                      const float* __restrict__ bias,
                      float* __restrict__ out0, float* __restrict__ out1,
                      int M, int N, int K)
{
    __shared__ __align__(16) _Float16 As[2][BM][BKP];
    __shared__ __align__(16) _Float16 Bs[2][BN][BKP];

    const int tid  = threadIdx.x;
    const int lane = tid & 31;
    const int wave = tid >> 5;
    const int wm   = wave >> 2;          // 0..1
    const int wn   = wave & 3;           // 0..3
    const int rowBase = blockIdx.y * BM;
    const int colBase = blockIdx.x * BN;
    const int m0 = wm * 64;
    const int n0 = wn * 32;
    const int hi  = lane >> 4;           // 0/1
    const int l15 = lane & 15;

    f32x8 acc[4][2];
#pragma unroll
    for (int i = 0; i < 4; i++)
#pragma unroll
        for (int j = 0; j < 2; j++)
#pragma unroll
            for (int e = 0; e < 8; e++) acc[i][j][e] = 0.0f;

    // Stage one BMxBK A-tile and BNxBK B-tile: 1024 16B chunks each -> 4 per thread.
    auto load_tiles = [&](int k0, int bufi) {
#pragma unroll
        for (int p = 0; p < 4; p++) {
            int chunk = p * 256 + tid;
            int r  = chunk >> 3;              // 8 chunks per 64-half row
            int kk = (chunk & 7) * 8;
            const _Float16* ga = A  + (size_t)(rowBase + r) * K + k0 + kk;
            const _Float16* gb = Bt + (size_t)(colBase + r) * K + k0 + kk;
#if USE_ASYNC_LDS
            __builtin_amdgcn_global_load_async_to_lds_b128(
                (AS_GLOBAL i32x4*)ga, (AS_LDS i32x4*)&As[bufi][r][kk], 0, 0);
            __builtin_amdgcn_global_load_async_to_lds_b128(
                (AS_GLOBAL i32x4*)gb, (AS_LDS i32x4*)&Bs[bufi][r][kk], 0, 0);
#else
            *(half8*)&As[bufi][r][kk] = *(const half8*)ga;
            *(half8*)&Bs[bufi][r][kk] = *(const half8*)gb;
#endif
        }
    };

    load_tiles(0, 0);                        // prologue: tile 0 in flight
    int buf = 0;

    for (int k0 = 0; k0 < K; k0 += BK) {
        if (k0 + BK < K) {
            load_tiles(k0 + BK, buf ^ 1);    // prefetch next tile into other buffer
#if USE_ASYNC_LDS
            __builtin_amdgcn_s_wait_asynccnt(8);  // oldest 8 (current tile) landed
#endif
        } else {
#if USE_ASYNC_LDS
            __builtin_amdgcn_s_wait_asynccnt(0);
#endif
        }
        __syncthreads();                     // all waves' current-tile data visible

#pragma unroll
        for (int ks = 0; ks < 2; ks++) {
            // Fragments per ISA wave32 16-bit layouts (A 16x32, B 32x16).
            half16 af[4], bf[2];
#pragma unroll
            for (int i = 0; i < 4; i++) {
                int r  = m0 + i * 16 + l15;
                int kb = ks * 32 + hi * 8;   // lanes<16: K 0..7/16..23 ; >=16: 8..15/24..31
                half8 lo8 = *(const half8*)&As[buf][r][kb];
                half8 hi8 = *(const half8*)&As[buf][r][kb + 16];
                af[i] = __builtin_shufflevector(lo8, hi8, 0,1,2,3,4,5,6,7,8,9,10,11,12,13,14,15);
            }
#pragma unroll
            for (int j = 0; j < 2; j++) {
                int n  = n0 + j * 16 + l15;
                int kb = ks * 32 + hi * 8;   // lanes<16: K 0..7/16..23 ; >=16: 8..15/24..31
                half8 b0 = *(const half8*)&Bs[buf][n][kb];
                half8 b1 = *(const half8*)&Bs[buf][n][kb + 16];
                bf[j] = __builtin_shufflevector(b0, b1, 0,1,2,3,4,5,6,7,8,9,10,11,12,13,14,15);
            }
#pragma unroll
            for (int i = 0; i < 4; i++)
#pragma unroll
                for (int j = 0; j < 2; j++)
                    acc[i][j] = __builtin_amdgcn_wmma_f32_16x16x32_f16(
                        false, af[i], false, bf[j], (short)0, acc[i][j], false, false);
        }
        __syncthreads();                     // buffer free before next prefetch lands
        buf ^= 1;
    }

    // Epilogue. C/D layout: VGPR e, lanes<16 -> M=e, N=lane; lanes>=16 -> M=8+e, N=lane-16.
#pragma unroll
    for (int i = 0; i < 4; i++)
#pragma unroll
        for (int j = 0; j < 2; j++)
#pragma unroll
            for (int e = 0; e < 8; e++) {
                int row = rowBase + m0 + i * 16 + hi * 8 + e;
                int col = colBase + n0 + j * 16 + l15;
                float val = acc[i][j][e] + bias[col];
                if (MODE == 0) {
                    out0[(size_t)row * N + col] = val;
                } else {
                    int slot = col % 3, c = col / 3;
                    if (slot == 0)      out0[(size_t)row * (N / 3) + c] = val;
                    else if (slot == 2) out1[(size_t)row * (N / 3) + c] = val;
                }
            }
}

// ---------------- depthwise 3-tap conv fused with (L,C)->(C,L) transpose ----------------
__global__ __launch_bounds__(256)
void shortconv_transpose_kernel(const float* __restrict__ vpre,   // (B,L,C)
                                const float* __restrict__ sw,     // (C,1,3)
                                const float* __restrict__ sb,     // (C)
                                float* __restrict__ vt)           // (B*C, L)
{
    __shared__ float tile[34][33];
    int l0 = blockIdx.x * 32, c0 = blockIdx.y * 32, b = blockIdx.z;
    int tx = threadIdx.x, ty = threadIdx.y;
    for (int r = ty; r < 34; r += 8) {
        int gl = l0 + r - 1;
        float v = 0.0f;
        if (gl >= 0 && gl < Ls) v = vpre[((size_t)(b * Ls + gl)) * Cs + c0 + tx];
        tile[r][tx] = v;
    }
    __syncthreads();
    int l = l0 + tx;
    for (int cc = ty; cc < 32; cc += 8) {
        int c = c0 + cc;
        float w0 = sw[c * 3 + 0], w1 = sw[c * 3 + 1], w2 = sw[c * 3 + 2];
        float val = w0 * tile[tx][cc] + w1 * tile[tx + 1][cc] + w2 * tile[tx + 2][cc] + sb[c];
        vt[((size_t)(b * Cs + c)) * Ls + l] = val;
    }
}

// ---------------- generic (R,C)->(C,R) fp32 transpose ----------------
__global__ __launch_bounds__(256)
void transpose_kernel(const float* __restrict__ in, float* __restrict__ out, int R, int C)
{
    __shared__ float tile[32][33];
    int r0 = blockIdx.x * 32, c0 = blockIdx.y * 32;
    for (int rr = threadIdx.y; rr < 32; rr += 8)
        tile[rr][threadIdx.x] = in[(size_t)(r0 + rr) * C + c0 + threadIdx.x];
    __syncthreads();
    for (int rr = threadIdx.y; rr < 32; rr += 8)
        out[(size_t)(c0 + rr) * R + r0 + threadIdx.x] = tile[threadIdx.x][rr];
}

// ---------------- implicit filter MLP: only odd coef columns (order i=1) ----------------
__device__ __forceinline__ float gelu_exact(float x) {
    return 0.5f * x * (1.0f + erff(x * 0.7071067811865476f));
}

__global__ __launch_bounds__(256)
void filter_mlp_kernel(const float* __restrict__ fw1, const float* __restrict__ fb1,
                       const float* __restrict__ fw2, const float* __restrict__ fb2,
                       const float* __restrict__ fw3, const float* __restrict__ fb3,
                       float* __restrict__ hout)                   // (L, C)
{
    __shared__ float h1[64];
    __shared__ float h2[128];
    int l = blockIdx.x;
    int t = threadIdx.x;
    float pos = (float)l / (float)(Ls - 1);
    if (t < 64) h1[t] = gelu_exact(pos * fw1[t] + fb1[t]);
    __syncthreads();
    if (t < 128) {
        float acc = fb2[t];
#pragma unroll 8
        for (int i = 0; i < 64; i++) acc += h1[i] * fw2[i * 128 + t];
        h2[t] = gelu_exact(acc);
    }
    __syncthreads();
    for (int c = t; c < Cs; c += 256) {
        int col = c * 2 + 1;                 // coefs[:, :, 1]
        float acc = fb3[col];
#pragma unroll 8
        for (int i = 0; i < 128; i++) acc += h2[i] * fw3[i * 2048 + col];
        hout[(size_t)l * Cs + c] = acc;
    }
}

// ---------------- FFT long conv: per (b,c) channel, 2-for-1 real FFT ----------------
__device__ __forceinline__ void fft_stages(float* re, float* im, int tid, int nthr, float sign)
{
#pragma unroll 1
    for (int s = 1; s <= LGN; s++) {
        int half = 1 << (s - 1);
        int m    = 1 << s;
        for (int t = tid; t < (Nfft >> 1); t += nthr) {
            int j    = t & (half - 1);
            int base = (t >> (s - 1)) << s;
            float ang = sign * (6.283185307179586f * (float)j / (float)m);
            float wi, wr;
            sincosf(ang, &wi, &wr);
            int i0 = base + j, i1 = i0 + half;
            float ur = re[i0], ui = im[i0];
            float xr = re[i1], xi = im[i1];
            float tr = wr * xr - wi * xi;
            float ti = wr * xi + wi * xr;
            re[i0] = ur + tr; im[i0] = ui + ti;
            re[i1] = ur - tr; im[i1] = ui - ti;
        }
        __syncthreads();
    }
}

__global__ __launch_bounds__(512)
void fft_conv_kernel(const float* __restrict__ vt,   // (B*C, L)
                     const float* __restrict__ ht,   // (C, L)
                     float* __restrict__ convt)      // (B*C, L)
{
    extern __shared__ float sm[];
    float* re = sm;
    float* im = sm + Nfft;

    const int tid = threadIdx.x, nthr = 512;
    const int bc  = blockIdx.x;
    const int c   = bc & (Cs - 1);
    const float* vrow = vt + (size_t)bc * Ls;
    const float* hrow = ht + (size_t)c  * Ls;

    // z[l] = v[l] + i*h[l], zero-padded to 2L; store bit-reversed for DIT.
    for (int l = tid; l < Nfft; l += nthr) {
        int r = __brev((unsigned)l) >> (32 - LGN);
        float vr = (l < Ls) ? vrow[l] : 0.0f;
        float hr = (l < Ls) ? hrow[l] : 0.0f;
        re[r] = vr; im[r] = hr;
    }
    __syncthreads();

    fft_stages(re, im, tid, nthr, -1.0f);            // forward FFT

    // Split Z into V,H (both real), P = V*H, rebuild Hermitian P.
    for (int k = tid; k <= (Nfft >> 1); k += nthr) {
        int nk = (Nfft - k) & (Nfft - 1);
        float zr0 = re[k],  zi0 = im[k];
        float zr1 = re[nk], zi1 = im[nk];
        float vr = 0.5f * (zr0 + zr1), vi = 0.5f * (zi0 - zi1);
        float hr = 0.5f * (zi0 + zi1), hi = -0.5f * (zr0 - zr1);
        float pr = vr * hr - vi * hi;
        float pi = vr * hi + vi * hr;
        re[k] = pr; im[k] = pi;
        if (nk != k) { re[nk] = pr; im[nk] = -pi; }
    }
    __syncthreads();

    // In-place bit-reversal permutation, then inverse FFT.
    for (int l = tid; l < Nfft; l += nthr) {
        int r = __brev((unsigned)l) >> (32 - LGN);
        if (r > l) {
            float a = re[l]; re[l] = re[r]; re[r] = a;
            float b = im[l]; im[l] = im[r]; im[r] = b;
        }
    }
    __syncthreads();

    fft_stages(re, im, tid, nthr, +1.0f);            // inverse (scaled below)

    const float inv = 1.0f / (float)Nfft;
    for (int l = tid; l < Ls; l += nthr)
        convt[(size_t)bc * Ls + l] = re[l] * inv;
}

// ---------------- gate by x2, transpose back to (B*L, C), pack f16 ----------------
__global__ __launch_bounds__(256)
void gate_pack_kernel(const float* __restrict__ convt,  // (B*C, L)
                      const float* __restrict__ x2,     // (B,L,C)
                      _Float16* __restrict__ g)         // (B*L, C)
{
    __shared__ float tile[32][33];
    int l0 = blockIdx.x * 32, c0 = blockIdx.y * 32, b = blockIdx.z;
    int tx = threadIdx.x, ty = threadIdx.y;
    for (int cc = ty; cc < 32; cc += 8)
        tile[cc][tx] = convt[((size_t)(b * Cs + c0 + cc)) * Ls + l0 + tx];
    __syncthreads();
    for (int ll = ty; ll < 32; ll += 8) {
        size_t idx = ((size_t)(b * Ls + l0 + ll)) * Cs + c0 + tx;
        g[idx] = (_Float16)(x2[idx] * tile[tx][ll]);
    }
}

// ---------------------------------------------------------------------------
extern "C" void kernel_launch(void* const* d_in, const int* in_sizes, int n_in,
                              void* d_out, int out_size, void* d_ws, size_t ws_size,
                              hipStream_t stream)
{
    const float* x    = (const float*)d_in[0];   // (B,L,C)
    const float* w1   = (const float*)d_in[1];   // (C, 3C)
    const float* b1   = (const float*)d_in[2];   // (3C)
    const float* sw   = (const float*)d_in[3];   // (C,1,3)
    const float* sb   = (const float*)d_in[4];   // (C)
    const float* fw1  = (const float*)d_in[5];
    const float* fb1  = (const float*)d_in[6];
    const float* fw2  = (const float*)d_in[7];
    const float* fb2  = (const float*)d_in[8];
    const float* fw3  = (const float*)d_in[9];
    const float* fb3  = (const float*)d_in[10];
    const float* wout = (const float*)d_in[11];  // (C, C)
    const float* bout = (const float*)d_in[12];  // (C)
    float* out = (float*)d_out;

    char* ws = (char*)d_ws;
    const size_t OFF_XH   = 0;                        // 32MB f16 (reused by g)
    const size_t OFF_W1T  = OFF_XH  + (size_t)Mrow * Cs * 2;        // 6MB  (3C, C) f16
    const size_t OFF_W2T  = OFF_W1T + (size_t)Cs * 3 * Cs * 2;      // 2MB  (C, C)  f16
    const size_t OFF_VPRE = OFF_W2T + (size_t)Cs * Cs * 2;          // 64MB (reused by conv_t)
    const size_t OFF_X2   = OFF_VPRE + (size_t)Mrow * Cs * 4;       // 64MB
    const size_t OFF_VT   = OFF_X2   + (size_t)Mrow * Cs * 4;       // 64MB
    const size_t OFF_HT   = OFF_VT   + (size_t)Mrow * Cs * 4;       // 32MB
    const size_t OFF_HLC  = OFF_HT   + (size_t)Ls * Cs * 4;         // 32MB

    _Float16* xh    = (_Float16*)(ws + OFF_XH);
    _Float16* w1t   = (_Float16*)(ws + OFF_W1T);
    _Float16* w2t   = (_Float16*)(ws + OFF_W2T);
    float*    vpre  = (float*)(ws + OFF_VPRE);
    float*    x2    = (float*)(ws + OFF_X2);
    float*    vt    = (float*)(ws + OFF_VT);
    float*    ht    = (float*)(ws + OFF_HT);
    float*    hlc   = (float*)(ws + OFF_HLC);
    float*    convt = (float*)(ws + OFF_VPRE);       // reuse vpre
    _Float16* g     = (_Float16*)(ws + OFF_XH);      // reuse xh

    // 1) fp16 conversions; weights transposed to (N, K) for contiguous staging
    { int n = Mrow * Cs; cvt_f16_kernel<<<(n + 255) / 256, 256, 0, stream>>>(x, xh, n); }
    {
        dim3 grid(Cs / 32, (3 * Cs) / 32);
        cvt_transpose_f16_kernel<<<grid, dim3(32, 8), 0, stream>>>(w1, w1t, Cs, 3 * Cs);
    }
    {
        dim3 grid(Cs / 32, Cs / 32);
        cvt_transpose_f16_kernel<<<grid, dim3(32, 8), 0, stream>>>(wout, w2t, Cs, Cs);
    }

    // 2) in_proj GEMM (WMMA), split epilogue -> v_pre, x2
    {
        dim3 grid((3 * Cs) / BN, Mrow / BM);
        wmma_gemm_kernel<1><<<grid, 256, 0, stream>>>(xh, w1t, b1, vpre, x2, Mrow, 3 * Cs, Cs);
    }

    // 3) depthwise conv + transpose -> v_t (B*C, L)
    {
        dim3 grid(Ls / 32, Cs / 32, Bb);
        shortconv_transpose_kernel<<<grid, dim3(32, 8), 0, stream>>>(vpre, sw, sb, vt);
    }

    // 4) implicit filter MLP -> h_lc (L, C), then transpose -> h_t (C, L)
    filter_mlp_kernel<<<Ls, 256, 0, stream>>>(fw1, fb1, fw2, fb2, fw3, fb3, hlc);
    {
        dim3 grid(Ls / 32, Cs / 32);
        transpose_kernel<<<grid, dim3(32, 8), 0, stream>>>(hlc, ht, Ls, Cs);
    }

    // 5) FFT long conv per (b,c): 2048 blocks, 128KB dynamic LDS each
    fft_conv_kernel<<<Bb * Cs, 512, 2 * Nfft * sizeof(float), stream>>>(vt, ht, convt);

    // 6) gate by x2, transpose to (B*L, C), pack f16
    {
        dim3 grid(Ls / 32, Cs / 32, Bb);
        gate_pack_kernel<<<grid, dim3(32, 8), 0, stream>>>(convt, x2, g);
    }

    // 7) out GEMM (WMMA) -> d_out
    {
        dim3 grid(Cs / BN, Mrow / BM);
        wmma_gemm_kernel<0><<<grid, 256, 0, stream>>>(g, w2t, bout, out, nullptr, Mrow, Cs, Cs);
    }
}
